// DSRB_19447611916345
// MI455X (gfx1250) — matrix operand order repaired and checked by reading, not compiled, not run on hardware
//
#include <hip/hip_runtime.h>
#include <hip/hip_bf16.h>
#include <stdint.h>

typedef __attribute__((ext_vector_type(16))) __bf16          v16bf;
typedef __attribute__((ext_vector_type(16))) unsigned short  v16u;
typedef __attribute__((ext_vector_type(8)))  float           v8f;

#define C_CH   256
#define HD     64
#define HW     128
#define CHW    (128 * 128)          // 16384
#define NPIX   (8LL * 256 * 128 * 128)
#define WFRAG_N (4 * 16 * 8 * 32 * 16)   // [shift][mtile][chunk][lane][elem]

// native f32 -> bf16 (RNE) via hardware cvt
__device__ __forceinline__ unsigned short f2bf(float f) {
    return __builtin_bit_cast(unsigned short, (__bf16)f);
}

// ---------------------------------------------------------------------------
// Pre-pass A: cast x to bf16 AND transpose to channel-last xt[b][h][w][c].
// Channel-last makes each 32-channel K-chunk for one (row,col) a contiguous
// 64-byte run -> async-copyable straight into the WMMA LDS layout.
// Vectorized: float4 loads -> b64 LDS stores -> ushort4 global stores.
// ---------------------------------------------------------------------------
__global__ __launch_bounds__(256)
void dsrb_xcast(const float* __restrict__ x, unsigned short* __restrict__ xt)
{
    __shared__ unsigned short ls[128][132];            // 66-dword rows, 8B-aligned
    const int b = blockIdx.x >> 7;
    const int h = blockIdx.x & 127;
    const float* xb = x + (long long)b * C_CH * CHW + h * HW;   // + c*CHW + w
    unsigned short* xto = xt + ((long long)(b * HW + h) * HW) * C_CH;  // + w*256 + c

    for (int half = 0; half < 2; ++half) {
        const int cb = half * 128;
        for (int i = threadIdx.x; i < 128 * 32; i += 256) {
            const int c = i >> 5, w4 = (i & 31) << 2;           // coalesced float4
            const float4 v = *(const float4*)&xb[(long long)(cb + c) * CHW + w4];
            ls[c][w4 + 0] = f2bf(v.x);
            ls[c][w4 + 1] = f2bf(v.y);
            ls[c][w4 + 2] = f2bf(v.z);
            ls[c][w4 + 3] = f2bf(v.w);
        }
        __syncthreads();
        for (int i = threadIdx.x; i < 128 * 32; i += 256) {
            const int w = i >> 5, c4 = (i & 31) << 2;           // coalesced ushort4
            ushort4 o;
            o.x = ls[c4 + 0][w];
            o.y = ls[c4 + 1][w];
            o.z = ls[c4 + 2][w];
            o.w = ls[c4 + 3][w];
            *(ushort4*)&xto[(long long)w * C_CH + cb + c4] = o;
        }
        __syncthreads();
    }
}

// ---------------------------------------------------------------------------
// Pre-pass B: swizzle w1..w4 into the exact A-fragment per-lane element order
// (16x32 bf16, lane M=lane%16, K groups {0..7,16..23}/{8..15,24..31}).
// Also zero-fills the 64B OOB pad.
// ---------------------------------------------------------------------------
__global__ void dsrb_wswizzle(const float* __restrict__ w1, const float* __restrict__ w2,
                              const float* __restrict__ w3, const float* __restrict__ w4,
                              unsigned short* __restrict__ wfrag,
                              unsigned short* __restrict__ zpad)
{
    const int idx = blockIdx.x * blockDim.x + threadIdx.x;
    if (blockIdx.x == 0 && threadIdx.x < 32) zpad[threadIdx.x] = 0;
    if (idx >= WFRAG_N) return;
    const int e    = idx & 15;
    const int lane = (idx >> 4) & 31;
    const int cc   = (idx >> 9) & 7;
    const int m    = (idx >> 12) & 15;
    const int s    = (idx >> 16) & 3;
    const float* wp = (s == 0) ? w1 : (s == 1) ? w2 : (s == 2) ? w3 : w4;
    const int o  = m * 16 + (lane & 15);
    const int kb = (lane & 16) ? 8 : 0;
    const int k  = kb + (e < 8 ? e : e + 8);
    wfrag[idx] = f2bf(wp[o * C_CH + cc * 32 + k]);
}

// ---------------------------------------------------------------------------
// Async staging of one K-chunk (32 channels, two halo rows, 132 cols) into one
// LDS double-buffer half.  352 threads x exactly 3 x 16B segments = 1056, so
// every thread has a uniform ASYNCcnt contribution -> s_wait_asynccnt 3 in the
// pipelined loop retires this chunk while the next chunk's copies fly.
// ---------------------------------------------------------------------------
__device__ __forceinline__
void dsrb_stage_async(const unsigned short* __restrict__ xt,
                      const unsigned short* __restrict__ zpad,
                      unsigned short (*__restrict__ buf)[132][32],
                      int b, int rtop, int rbot, int c0, int tid)
{
    if (tid < 352) {
        #pragma unroll
        for (int j = 0; j < 3; ++j) {
            const int sgi = tid * 3 + j;                // 0..1055
            const int r   = sgi / 528;
            const int rem = sgi % 528;
            const int col = rem >> 2;
            const int q   = rem & 3;
            const int row = (r == 0) ? rtop : rbot;
            const int wg  = col - 2;
            const unsigned short* src =
                ((unsigned)row < 128u && (unsigned)wg < 128u)
                    ? (xt + (((long long)(b * HW + row) * HW + wg) * C_CH + c0 + q * 8))
                    : zpad;
            const unsigned long long ga   = (unsigned long long)(uintptr_t)src;
            const unsigned int       ldsa =
                (unsigned int)(unsigned long long)(uintptr_t)&buf[r][col][q * 8];
            asm volatile("global_load_async_to_lds_b128 %0, %1, off"
                         :: "v"(ldsa), "v"(ga) : "memory");
        }
    }
}

// ---------------------------------------------------------------------------
// Kernel 1: fused 4x shifted 1x1 conv (bf16 WMMA, K=1024) + residual sigmoid
// gate + relu.  Double-buffered GLOBAL_LOAD_ASYNC_TO_LDS_B128 staging from the
// channel-last bf16 copy; each chunk's DMA overlaps the previous chunk's WMMAs.
// ---------------------------------------------------------------------------
__global__ __launch_bounds__(512)
void dsrb_main_kernel(const float* __restrict__ x,
                      const unsigned short* __restrict__ xt,
                      const unsigned short* __restrict__ wfrag,
                      const unsigned short* __restrict__ zpad,
                      const float* __restrict__ b1, const float* __restrict__ b2,
                      const float* __restrict__ b3, const float* __restrict__ b4,
                      float* __restrict__ out,
                      float* __restrict__ partial)
{
    __shared__ unsigned short xs[2][2][132][32];   // [dbuf][row(top/bot)][col][k]
    __shared__ float bsum[C_CH];

    const int b    = blockIdx.x >> 7;
    const int h    = blockIdx.x & 127;
    const int tid  = threadIdx.x;
    const int wave = tid >> 5;
    const int lane = tid & 31;
    const int mb   = wave * 16;

    if (tid < C_CH) bsum[tid] = b1[tid] + b2[tid] + b3[tid] + b4[tid];

    const float* xb = x + (long long)b * C_CH * CHW;
    const v16u*  wfp = (const v16u*)wfrag;

    v8f acc[8];
    #pragma unroll
    for (int t = 0; t < 8; ++t)
        #pragma unroll
        for (int e = 0; e < 8; ++e) acc[t][e] = 0.0f;

    const int kbB  = (lane & 16) ? 8 : 0;
    const int rtop = h - 2, rbot = h + 2;

    // prologue: stage chunk 0 into buffer 0
    dsrb_stage_async(xt, zpad, xs[0], b, rtop, rbot, 0, tid);

    for (int cc = 0; cc < 8; ++cc) {
        // issue next chunk into the other buffer, then retire current chunk:
        // async loads complete in order, so asynccnt<=3 (the next chunk's
        // in-flight copies) implies this chunk's copies have landed.
        if (cc < 7) {
            dsrb_stage_async(xt, zpad, xs[(cc + 1) & 1], b, rtop, rbot,
                             (cc + 1) * 32, tid);
            asm volatile("s_wait_asynccnt 0x3" ::: "memory");
        } else {
            asm volatile("s_wait_asynccnt 0x0" ::: "memory");
        }
        __syncthreads();

        const unsigned short (*__restrict__ cbuf)[132][32] =
            (const unsigned short (*)[132][32])xs[cc & 1];

        // ---- 4 shifts x 8 N-tiles of v_wmma_f32_16x16x32_bf16 -------------
        #pragma unroll
        for (int s = 0; s < 4; ++s) {
            const v16u  au = wfp[((s * 16 + wave) * 8 + cc) * 32 + lane];
            const v16bf af = __builtin_bit_cast(v16bf, au);
            const int   r   = s & 1;            // s0,s2: top row  s1,s3: bottom
            const int   off = (s >= 2) ? 4 : 0; // col shift -2 vs +2

            #pragma unroll
            for (int t = 0; t < 8; ++t) {
                const int col = t * 16 + (lane & 15) + off;
                v16u bu;
                #pragma unroll
                for (int e = 0; e < 16; ++e) bu[e] = cbuf[r][col][kbB + (e < 8 ? e : e + 8)];
                const v16bf bfr = __builtin_bit_cast(v16bf, bu);
                acc[t] = __builtin_amdgcn_wmma_f32_16x16x32_bf16(
                    false, af, false, bfr, (short)0, acc[t], false, false);
            }
        }
        __syncthreads();   // buffer (cc&1) free for reuse at cc+2's issue
    }

    // ---- epilogue: bias, residual sigmoid gate, relu, pool partials -------
    const int n       = lane & 15;
    const int halfsel = (lane >> 4) & 1;
    float csum[8];
    #pragma unroll
    for (int j = 0; j < 8; ++j) csum[j] = 0.0f;

    #pragma unroll
    for (int t = 0; t < 8; ++t) {
        const int w = t * 16 + n;
        #pragma unroll
        for (int j = 0; j < 8; ++j) {
            const int   o  = mb + j + halfsel * 8;
            const float sv = acc[t][j] + bsum[o];
            const float xv = xb[(long long)o * CHW + h * HW + w];
            const float rg = 2.0f / (1.0f + __expf(-(xv - 0.25f * sv))) - 1.0f;
            const float hv = fmaxf(xv * rg, 0.0f);
            out[((long long)(b * C_CH + o) * HW + h) * HW + w] = hv;
            csum[j] += hv;
        }
    }
    #pragma unroll
    for (int j = 0; j < 8; ++j) {
        float v = csum[j];
        v += __shfl_xor(v, 1, 32);
        v += __shfl_xor(v, 2, 32);
        v += __shfl_xor(v, 4, 32);
        v += __shfl_xor(v, 8, 32);
        if (n == 0) {
            const int o = mb + j + halfsel * 8;
            partial[(long long)(b * HW + h) * C_CH + o] = v;
        }
    }
}

// ---------------------------------------------------------------------------
// Kernel 2: deterministic reduce of pool partials over h -> sums[b*256+c]
// ---------------------------------------------------------------------------
__global__ void dsrb_pool_reduce(const float* __restrict__ partial,
                                 float* __restrict__ sums)
{
    const int idx = blockIdx.x * blockDim.x + threadIdx.x;
    if (idx >= 8 * C_CH) return;
    const int b = idx >> 8, c = idx & 255;
    float s = 0.0f;
    for (int hh = 0; hh < HW; ++hh)
        s += partial[(long long)(b * HW + hh) * C_CH + c];
    sums[idx] = s;
}

// ---------------------------------------------------------------------------
// Kernel 3: AGCA channel-attention chain -> gate[b*256+c]
// ---------------------------------------------------------------------------
__global__ void dsrb_agca(const float* __restrict__ sums,
                          const float* __restrict__ aw1,
                          const float* __restrict__ aw2,
                          const float* __restrict__ aw3,
                          const float* __restrict__ aA2,
                          const float* __restrict__ aw4,
                          float* __restrict__ gate)
{
    const int b = threadIdx.x;
    if (b >= 8) return;
    const float* ys = sums + b * C_CH;
    float y1[HD];
    for (int j = 0; j < HD; ++j) {
        float a = 0.0f;
        for (int c = 0; c < C_CH; ++c)
            a += aw1[j * C_CH + c] * (ys[c] * (1.0f / (float)CHW));
        y1[j] = a;
    }
    const float w2v = aw2[0], w3v = aw3[0];
    float y3[HD];
    for (int k = 0; k < HD; ++k) {
        const float a1k = 1.0f / (1.0f + __expf(-w2v * y1[k]));
        float a = y1[k] * a1k;                       // diag(a1) term (A0 = I)
        for (int j = 0; j < HD; ++j) a += y1[j] * aA2[j * HD + k];
        y3[k] = fmaxf(w3v * a, 0.0f);
    }
    for (int c = 0; c < C_CH; ++c) {
        float g = 0.0f;
        for (int k = 0; k < HD; ++k) g += y3[k] * aw4[c * HD + k];
        gate[b * C_CH + c] = 1.0f / (1.0f + __expf(-g));
    }
}

// ---------------------------------------------------------------------------
// Kernel 4: in-place channel-gated scale of h (float4)
// ---------------------------------------------------------------------------
__global__ void dsrb_scale(float* __restrict__ out, const float* __restrict__ gate)
{
    const long long i = ((long long)blockIdx.x * blockDim.x + threadIdx.x) * 4;
    if (i >= NPIX) return;
    const int bc = (int)(i >> 14);                   // b*256+c (16384 = H*W)
    const float g = gate[bc];
    float4 v = *(const float4*)(out + i);
    v.x *= g; v.y *= g; v.z *= g; v.w *= g;
    *(float4*)(out + i) = v;
}

extern "C" void kernel_launch(void* const* d_in, const int* in_sizes, int n_in,
                              void* d_out, int out_size, void* d_ws, size_t ws_size,
                              hipStream_t stream) {
    (void)in_sizes; (void)n_in; (void)out_size; (void)ws_size;
    const float* x   = (const float*)d_in[0];
    const float* w1  = (const float*)d_in[1];
    const float* b1  = (const float*)d_in[2];
    const float* w2  = (const float*)d_in[3];
    const float* b2  = (const float*)d_in[4];
    const float* w3  = (const float*)d_in[5];
    const float* b3  = (const float*)d_in[6];
    const float* w4  = (const float*)d_in[7];
    const float* b4  = (const float*)d_in[8];
    const float* aw1 = (const float*)d_in[9];
    const float* aw2 = (const float*)d_in[10];
    const float* aw3 = (const float*)d_in[11];
    const float* aA2 = (const float*)d_in[12];
    const float* aw4 = (const float*)d_in[13];
    float* out = (float*)d_out;

    // workspace layout (bytes)
    char* wsb = (char*)d_ws;
    unsigned short* xt    = (unsigned short*)(wsb);                      // 67,108,864 B
    unsigned short* wfrag = (unsigned short*)(wsb + 67108864);           //    524,288 B
    unsigned short* zpad  = (unsigned short*)(wsb + 67633152);           //        128 B
    float* partial        = (float*)(wsb + 67633280);                    //  1,048,576 B
    float* sums           = (float*)(wsb + 68681856);                    //      8,192 B
    float* gate           = (float*)(wsb + 68690048);                    //      8,192 B

    dsrb_xcast<<<8 * 128, 256, 0, stream>>>(x, xt);
    dsrb_wswizzle<<<WFRAG_N / 256, 256, 0, stream>>>(w1, w2, w3, w4, wfrag, zpad);
    dsrb_main_kernel<<<8 * 128, 512, 0, stream>>>(x, xt, wfrag, zpad,
                                                  b1, b2, b3, b4, out, partial);
    dsrb_pool_reduce<<<8, 256, 0, stream>>>(partial, sums);
    dsrb_agca<<<1, 32, 0, stream>>>(sums, aw1, aw2, aw3, aA2, aw4, gate);
    const long long nvec = NPIX / 4;
    dsrb_scale<<<(int)((nvec + 255) / 256), 256, 0, stream>>>(out, gate);
}
